// Mamba2_74904229642380
// MI455X (gfx1250) — matrix operand order repaired and checked by reading, compiled, not run on hardware
//
#include <hip/hip_runtime.h>
#include <hip/hip_bf16.h>
#include <cstdint>
#include <cstddef>

// Problem dimensions (fixed by the reference)
#define DM     1024     // d_model
#define DSTATE 16       // d_state
#define DCONV  4        // d_conv
#define DI     2048     // d_inner
#define DTR    64       // dt_rank
#define BSZ    2
#define LSEQ   2048
#define BL     (BSZ * LSEQ)   // 4096 rows for all GEMMs

typedef __attribute__((ext_vector_type(16))) __bf16 v16bf;
typedef __attribute__((ext_vector_type(8)))  __bf16 v8bf;
typedef __attribute__((ext_vector_type(8)))  float  v8f;

// ---------------------------------------------------------------------------
// Fragment loaders per CDNA5 16-bit WMMA VGPR layouts (wave32).
// A (16x32, MxK): lane_lo = row M; sel = lane>>4.
//   elems 0..7  = K[sel*8 .. sel*8+7], elems 8..15 = K[sel*8+16 .. sel*8+23]
// B (32x16, KxN): lane_lo = col N; same K order as A
// Both A and B are stored K-contiguous in memory (NT gemm), so loads are
// 16-byte b128 transfers.
// ---------------------------------------------------------------------------
__device__ __forceinline__ v16bf load_frag_a(const __bf16* p) {
  v8bf lo = *reinterpret_cast<const v8bf*>(p);
  v8bf hi = *reinterpret_cast<const v8bf*>(p + 16);
  v16bf r;
#pragma unroll
  for (int i = 0; i < 8; ++i) { r[i] = lo[i]; r[i + 8] = hi[i]; }
  return r;
}
__device__ __forceinline__ v16bf load_frag_b(const __bf16* p) {
  v8bf lo = *reinterpret_cast<const v8bf*>(p);
  v8bf hi = *reinterpret_cast<const v8bf*>(p + 16);
  v16bf r;
#pragma unroll
  for (int i = 0; i < 8; ++i) { r[i] = lo[i]; r[i + 8] = hi[i]; }
  return r;
}

// ---------------------------------------------------------------------------
// NT-GEMM: C[M,N] (f32) = A[M,K] (bf16, row-major) x B[N,K]^T (bf16, row-major)
// One wave -> 16 x (16*NT) output tile; A fragment reused across NT WMMAs.
// Block = 256 threads = 8 waves stacked along M (block tile 128 x 16*NT).
// Requires M % 128 == 0, N % (16*NT) == 0, K % 32 == 0 (true for all calls).
// ---------------------------------------------------------------------------
template <int NT>
__global__ void __launch_bounds__(256)
gemm_nt_bf16f32(const __bf16* __restrict__ A, const __bf16* __restrict__ B,
                float* __restrict__ C, int M, int N, int K) {
  const int lane    = threadIdx.x & 31;
  const int wave    = threadIdx.x >> 5;
  const int lane_lo = lane & 15;
  const int sel     = lane >> 4;                 // 0 or 1 (lane half)
  const int tile_m  = (blockIdx.y * 8 + wave) * 16;
  const int tile_n  = blockIdx.x * (16 * NT);
  if (tile_m >= M) return;                       // wave-uniform guard

  v8f acc[NT];
#pragma unroll
  for (int t = 0; t < NT; ++t)
#pragma unroll
    for (int i = 0; i < 8; ++i) acc[t][i] = 0.0f;

  const __bf16* arow = A + (size_t)(tile_m + lane_lo) * K + sel * 8;
  for (int k0 = 0; k0 < K; k0 += 32) {
    v16bf afrag = load_frag_a(arow + k0);
#pragma unroll
    for (int t = 0; t < NT; ++t) {
      const __bf16* brow =
          B + (size_t)(tile_n + t * 16 + lane_lo) * K + k0 + sel * 8;
      v16bf bfrag = load_frag_b(brow);
      acc[t] = __builtin_amdgcn_wmma_f32_16x16x32_bf16(
          /*neg_a=*/false, afrag, /*neg_b=*/false, bfrag,
          /*c_mod=*/(short)0, acc[t], /*reuse_a=*/false, /*reuse_b=*/false);
    }
  }

  // C/D layout: VGPR i holds (M = i + sel*8, N = lane_lo)
#pragma unroll
  for (int t = 0; t < NT; ++t) {
    float* crow = C + (size_t)(tile_m + sel * 8) * N + tile_n + t * 16 + lane_lo;
#pragma unroll
    for (int i = 0; i < 8; ++i) crow[(size_t)i * N] = acc[t][i];
  }
}

// ---------------------------------------------------------------------------
// Elementwise helpers
// ---------------------------------------------------------------------------
__global__ void cvt_f32_bf16(const float* __restrict__ in,
                             __bf16* __restrict__ out, int n) {
  int i = blockIdx.x * blockDim.x + threadIdx.x;
  if (i < n) out[i] = (__bf16)in[i];
}

// Causal depthwise conv (d_conv=4) + SiLU. Reads x_in = xz[:, :, 0:DI].
// Writes f32 (scan input) and bf16 (GEMM operand) copies.
__global__ void __launch_bounds__(256)
conv_silu(const float* __restrict__ xz, const float* __restrict__ w,
          const float* __restrict__ bias, float* __restrict__ xc_f,
          __bf16* __restrict__ xc_bf) {
  int tid = blockIdx.x * blockDim.x + threadIdx.x;     // over BL*DI
  int d = tid & (DI - 1);
  int m = tid >> 11;                                    // b*LSEQ + l
  int l = m & (LSEQ - 1);
  float s = bias[d];
#pragma unroll
  for (int j = 0; j < DCONV; ++j) {
    int lj = l - (DCONV - 1) + j;
    if (lj >= 0)
      s += w[d * DCONV + j] * xz[(size_t)(m - (DCONV - 1) + j) * (2 * DI) + d];
  }
  float v = s / (1.0f + expf(-s));                      // SiLU
  xc_f[tid]  = v;
  xc_bf[tid] = (__bf16)v;
}

// dt = softplus(dt_lin + dt_bias), in place over (BL, DI)
__global__ void __launch_bounds__(256)
softplus_bias(float* __restrict__ dt, const float* __restrict__ bias) {
  int tid = blockIdx.x * blockDim.x + threadIdx.x;
  int d = tid & (DI - 1);
  float v = dt[tid] + bias[d];
  dt[tid] = (v > 20.0f) ? v : log1pf(expf(v));
}

// ---------------------------------------------------------------------------
// Selective scan: one thread per (b, d_inner) channel, 16-wide state in regs.
// Fuses y += D*x ; y *= silu(z) ; bf16 conversion for out_proj.
// ---------------------------------------------------------------------------
__global__ void __launch_bounds__(256)
selective_scan(const float* __restrict__ xc, const float* __restrict__ dt,
               const float* __restrict__ A_log, const float* __restrict__ Dvec,
               const float* __restrict__ Bm, const float* __restrict__ Cm,
               const float* __restrict__ xz, __bf16* __restrict__ ybf) {
  int tid = blockIdx.x * blockDim.x + threadIdx.x;     // 0 .. BL-1 (4096)
  int d = tid & (DI - 1);
  int b = tid >> 11;
  float h[DSTATE];
  float Av[DSTATE];
#pragma unroll
  for (int s = 0; s < DSTATE; ++s) {
    h[s]  = 0.0f;
    Av[s] = -expf(A_log[d * DSTATE + s]);
  }
  float Dd = Dvec[d];
  for (int l = 0; l < LSEQ; ++l) {
    size_t m  = (size_t)b * LSEQ + l;
    float xv  = xc[m * DI + d];
    float dtv = dt[m * DI + d];
    float y = 0.0f;
#pragma unroll
    for (int s = 0; s < DSTATE; ++s) {
      float hs = h[s];
      hs += dtv * (Av[s] * hs) + dtv * (Bm[m * DSTATE + s] * xv);
      h[s] = hs;
      y += hs * Cm[m * DSTATE + s];
    }
    y += Dd * xv;
    float zv = xz[m * (2 * DI) + DI + d];
    float out = y * (zv / (1.0f + expf(-zv)));
    ybf[m * DI + d] = (__bf16)out;
  }
}

// ---------------------------------------------------------------------------
// Host-side orchestration
// ---------------------------------------------------------------------------
extern "C" void kernel_launch(void* const* d_in, const int* in_sizes, int n_in,
                              void* d_out, int out_size, void* d_ws, size_t ws_size,
                              hipStream_t stream) {
  const float* x      = (const float*)d_in[0];
  const float* in_w   = (const float*)d_in[1];
  const float* conv_w = (const float*)d_in[2];
  const float* conv_b = (const float*)d_in[3];
  const float* A_log  = (const float*)d_in[4];
  const float* Dvec   = (const float*)d_in[5];
  const float* dt_w   = (const float*)d_in[6];
  const float* dt_b   = (const float*)d_in[7];
  const float* xp_w   = (const float*)d_in[8];
  const float* Bp_w   = (const float*)d_in[9];
  const float* Cp_w   = (const float*)d_in[10];
  const float* out_w  = (const float*)d_in[11];

  char* base = (char*)d_ws;
  size_t off = 0;
  auto alloc = [&](size_t bytes) -> void* {
    void* p = base + off;
    off = (off + bytes + 255) & ~(size_t)255;
    return p;
  };
  __bf16* x_bf   = (__bf16*)alloc((size_t)BL * DM * 2);
  __bf16* winbf  = (__bf16*)alloc((size_t)2 * DI * DM * 2);
  __bf16* xpwbf  = (__bf16*)alloc((size_t)DTR * DI * 2);
  __bf16* dtwbf  = (__bf16*)alloc((size_t)DI * DTR * 2);
  __bf16* bpwbf  = (__bf16*)alloc((size_t)DSTATE * DI * 2);
  __bf16* cpwbf  = (__bf16*)alloc((size_t)DSTATE * DI * 2);
  __bf16* outwbf = (__bf16*)alloc((size_t)DM * DI * 2);
  float*  xz     = (float*)alloc((size_t)BL * 2 * DI * 4);
  float*  xc_f   = (float*)alloc((size_t)BL * DI * 4);
  __bf16* xc_bf  = (__bf16*)alloc((size_t)BL * DI * 2);
  float*  xp_f   = (float*)alloc((size_t)BL * DTR * 4);
  __bf16* xp_bf  = (__bf16*)alloc((size_t)BL * DTR * 2);
  float*  dtbuf  = (float*)alloc((size_t)BL * DI * 4);
  float*  Bmat   = (float*)alloc((size_t)BL * DSTATE * 4);
  float*  Cmat   = (float*)alloc((size_t)BL * DSTATE * 4);
  __bf16* ybf    = (__bf16*)alloc((size_t)BL * DI * 2);

  dim3 blk(256);
  auto cvt = [&](const float* in, __bf16* outp, int n) {
    cvt_f32_bf16<<<(n + 255) / 256, blk, 0, stream>>>(in, outp, n);
  };

  // 1) fp32 -> bf16 staging of activations + weights
  cvt(x, x_bf, BL * DM);
  cvt(in_w, winbf, 2 * DI * DM);
  cvt(xp_w, xpwbf, DTR * DI);
  cvt(dt_w, dtwbf, DI * DTR);
  cvt(Bp_w, bpwbf, DSTATE * DI);
  cvt(Cp_w, cpwbf, DSTATE * DI);
  cvt(out_w, outwbf, DM * DI);

  // 2) xz = x @ in_proj_w^T           (M=4096, N=4096, K=1024)
  gemm_nt_bf16f32<4><<<dim3((2 * DI) / 64, BL / 128), blk, 0, stream>>>(
      x_bf, winbf, xz, BL, 2 * DI, DM);

  // 3) causal depthwise conv + SiLU on xz[..., :DI]
  conv_silu<<<(BL * DI) / 256, blk, 0, stream>>>(xz, conv_w, conv_b, xc_f, xc_bf);

  // 4) x_proj: (BL x 64) = x_conv @ x_proj_w^T   (K=2048)
  gemm_nt_bf16f32<4><<<dim3(DTR / 64, BL / 128), blk, 0, stream>>>(
      xc_bf, xpwbf, xp_f, BL, DTR, DI);
  cvt(xp_f, xp_bf, BL * DTR);

  // 5) dt_lin: (BL x 2048) = xproj @ dt_proj_w^T  (K=64), then softplus+bias
  gemm_nt_bf16f32<4><<<dim3(DI / 64, BL / 128), blk, 0, stream>>>(
      xp_bf, dtwbf, dtbuf, BL, DI, DTR);
  softplus_bias<<<(BL * DI) / 256, blk, 0, stream>>>(dtbuf, dt_b);

  // 6) B, C projections: (BL x 16), N=16 -> single-tile variant
  gemm_nt_bf16f32<1><<<dim3(1, BL / 128), blk, 0, stream>>>(
      xc_bf, bpwbf, Bmat, BL, DSTATE, DI);
  gemm_nt_bf16f32<1><<<dim3(1, BL / 128), blk, 0, stream>>>(
      xc_bf, cpwbf, Cmat, BL, DSTATE, DI);

  // 7) sequential selective scan (+ D*x, * silu(z), bf16 out)
  selective_scan<<<dim3(BL / 256), blk, 0, stream>>>(
      xc_f, dtbuf, A_log, Dvec, Bmat, Cmat, xz, ybf);

  // 8) out = y @ out_proj_w^T  (M=4096, N=1024, K=2048) -> f32 d_out
  gemm_nt_bf16f32<4><<<dim3(DM / 64, BL / 128), blk, 0, stream>>>(
      ybf, outwbf, (float*)d_out, BL, DM, DI);

  (void)in_sizes; (void)n_in; (void)out_size; (void)ws_size;
}
